// LiquidLM_73143293051250
// MI455X (gfx1250) — compile-verified, matrix operand order checked by reading, not run on hardware
//
#include <hip/hip_runtime.h>
#include <hip/hip_bf16.h>
#include <math.h>

// ---------------------------------------------------------------------------
// Model dims (match reference)
#define VV    32000
#define DD    768
#define LL    12
#define RR    128
#define SS    64
#define NHH   8
#define HD    96      // DD / NHH
#define HALF  48
#define BBB   2
#define TT    2048
#define MTOT  (BBB*TT)        // 4096 rows
#define FF    (4*DD)          // 3072

typedef __attribute__((ext_vector_type(16))) __bf16 bf16x16;
typedef __attribute__((ext_vector_type(8)))  __bf16 bf16x8;
typedef __attribute__((ext_vector_type(8)))  float  f32x8;

// ---------------------------------------------------------------------------
// Convert 8 contiguous fp32 -> 8 bf16 and store 16B to LDS (ds_store_b128).
__device__ __forceinline__ void stage8(const float* __restrict__ src,
                                       __bf16* __restrict__ dst) {
  float4 a = *(const float4*)(src);
  float4 b = *(const float4*)(src + 4);
  bf16x8 v;
  v[0] = (__bf16)a.x; v[1] = (__bf16)a.y; v[2] = (__bf16)a.z; v[3] = (__bf16)a.w;
  v[4] = (__bf16)b.x; v[5] = (__bf16)b.y; v[6] = (__bf16)b.z; v[7] = (__bf16)b.w;
  *(bf16x8*)dst = v;
}

// ---------------------------------------------------------------------------
// WMMA GEMM:  C[M,N] = A[M,K] @ W[N,K]^T (+bias[n])
// block = 128 threads (4 wave32).  Block tile: 64(M) x 32(N).
// Each wave owns one 16-row strip and two 16x16 C tiles (2 accumulators,
// 2 v_wmma per K-step).  fp32 operands staged to LDS as bf16, double-buffered.
// Requires M%64==0, N%32==0, K%32==0 (true for all calls here).
__global__ __launch_bounds__(128)
void wmma_gemm_bf16(const float* __restrict__ A, const float* __restrict__ W,
                    const float* __restrict__ bias, float* __restrict__ C,
                    int M, int N, int K) {
  __shared__ __bf16 As[2][64 * 32];
  __shared__ __bf16 Ws[2][32 * 32];
  const int tid  = threadIdx.x;
  const int lane = tid & 31;
  const int wid  = tid >> 5;          // 0..3
  const int bm   = blockIdx.x * 64;
  const int bn   = blockIdx.y * 32;
  const int half = lane >> 4;         // 0|1
  const int lrow = lane & 15;

  // staging assignment: 8-float chunks
  const int ar0 = tid >> 2;                 // A rows for chunk 0 (0..31)
  const int ac  = (tid & 3) << 3;           // col within 32 (0,8,16,24)
  const int wr  = tid >> 2;                 // W row (0..31)

  f32x8 acc0 = {}, acc1 = {};

  // prologue: stage k-tile 0 into buffer 0
  {
    stage8(&A[(size_t)(bm + ar0) * K + ac],        &As[0][ar0 * 32 + ac]);
    stage8(&A[(size_t)(bm + ar0 + 32) * K + ac],   &As[0][(ar0 + 32) * 32 + ac]);
    stage8(&W[(size_t)(bn + wr) * K + ac],         &Ws[0][wr * 32 + ac]);
  }
  __syncthreads();

  int p = 0;
  for (int k0 = 0; k0 < K; k0 += 32) {
    const int np = p ^ 1;
    if (k0 + 32 < K) {
      const int kb = k0 + 32;
      stage8(&A[(size_t)(bm + ar0) * K + kb + ac],      &As[np][ar0 * 32 + ac]);
      stage8(&A[(size_t)(bm + ar0 + 32) * K + kb + ac], &As[np][(ar0 + 32) * 32 + ac]);
      stage8(&W[(size_t)(bn + wr) * K + kb + ac],       &Ws[np][wr * 32 + ac]);
      if (kb + 32 < K)  // pull the tile after next into cache (global_prefetch_b8)
        __builtin_prefetch(&A[(size_t)(bm + (tid & 63)) * K + (kb + 32)], 0, 1);
    }

    // fragments from buffer p (ISA 16-bit layout: e -> k=(e/8)*16+half*8+(e%8))
    const __bf16* ap = &As[p][(wid * 16 + lrow) * 32 + half * 8];
    bf16x8 a0 = *(const bf16x8*)ap;
    bf16x8 a1 = *(const bf16x8*)(ap + 16);
    bf16x16 af = __builtin_shufflevector(a0, a1, 0,1,2,3,4,5,6,7,8,9,10,11,12,13,14,15);

    const __bf16* bp0 = &Ws[p][lrow * 32 + half * 8];
    bf16x8 b0l = *(const bf16x8*)bp0;
    bf16x8 b0h = *(const bf16x8*)(bp0 + 16);
    bf16x16 bf0 = __builtin_shufflevector(b0l, b0h, 0,1,2,3,4,5,6,7,8,9,10,11,12,13,14,15);

    const __bf16* bp1 = &Ws[p][(16 + lrow) * 32 + half * 8];
    bf16x8 b1l = *(const bf16x8*)bp1;
    bf16x8 b1h = *(const bf16x8*)(bp1 + 16);
    bf16x16 bf1 = __builtin_shufflevector(b1l, b1h, 0,1,2,3,4,5,6,7,8,9,10,11,12,13,14,15);

    acc0 = __builtin_amdgcn_wmma_f32_16x16x32_bf16(false, af, false, bf0,
                                                   (short)0, acc0, false, false);
    acc1 = __builtin_amdgcn_wmma_f32_16x16x32_bf16(false, af, false, bf1,
                                                   (short)0, acc1, false, false);
    __syncthreads();
    p = np;
  }

  // C layout: VGPR r -> M = r + 8*half, N = lane%16
  const float bv0 = bias ? bias[bn + lrow] : 0.0f;
  const float bv1 = bias ? bias[bn + 16 + lrow] : 0.0f;
  #pragma unroll
  for (int r = 0; r < 8; ++r) {
    int m = bm + wid * 16 + half * 8 + r;
    C[(size_t)m * N + (bn + lrow)]      = acc0[r] + bv0;
    C[(size_t)m * N + (bn + 16 + lrow)] = acc1[r] + bv1;
  }
}

// ---------------------------------------------------------------------------
// Small helper kernels
__global__ void embed_gather(const int* __restrict__ ids,
                             const float* __restrict__ emb,
                             float* __restrict__ x, int n) {
  int i = blockIdx.x * blockDim.x + threadIdx.x;
  if (i >= n) return;
  int bt = i / DD, d = i % DD;
  x[i] = emb[(size_t)ids[bt] * DD + d];
}

__global__ void phi_elu1(float* x, int n) {           // elu(x)+1
  int i = blockIdx.x * blockDim.x + threadIdx.x;
  if (i >= n) return;
  float v = x[i];
  x[i] = v > 0.0f ? v + 1.0f : __expf(v);
}

__global__ void rope_k(float* x) {                    // in-place, [B*T, NH*HD]
  int i = blockIdx.x * blockDim.x + threadIdx.x;
  if (i >= MTOT * NHH * HALF) return;
  int j  = i % HALF;
  int h  = (i / HALF) % NHH;
  int bt = i / (HALF * NHH);
  int t  = bt % TT;
  float fr  = __powf(10000.0f, -(float)j / (float)HALF);
  float ang = (float)t * fr;
  float sn = __sinf(ang), cs = __cosf(ang);
  size_t base = (size_t)bt * DD + h * HD;
  float x1 = x[base + j], x2 = x[base + HALF + j];
  x[base + j]        = x1 * cs - x2 * sn;
  x[base + HALF + j] = x1 * sn + x2 * cs;
}

__global__ void kv_outer(const float* __restrict__ k, const float* __restrict__ v,
                         float* __restrict__ kv) {    // kv[b,h,d,e]
  int i = blockIdx.x * blockDim.x + threadIdx.x;
  if (i >= BBB * NHH * HD * HD) return;
  int e = i % HD, d = (i / HD) % HD, h = (i / (HD * HD)) % NHH, b = i / (HD * HD * NHH);
  float acc = 0.0f;
  for (int t = 0; t < TT; ++t) {
    size_t o = ((size_t)(b * TT + t)) * DD + h * HD;
    acc += k[o + d] * v[o + e];
  }
  kv[i] = acc;
}

__global__ void k_sum(const float* __restrict__ k, float* __restrict__ ks) {
  int i = blockIdx.x * blockDim.x + threadIdx.x;       // b*NH*HD
  if (i >= BBB * NHH * HD) return;
  int d = i % HD, h = (i / HD) % NHH, b = i / (HD * NHH);
  float acc = 0.0f;
  for (int t = 0; t < TT; ++t)
    acc += k[((size_t)(b * TT + t)) * DD + h * HD + d];
  ks[i] = acc;
}

__global__ void z_denom(const float* __restrict__ q, const float* __restrict__ ks,
                        float* __restrict__ z) {       // z[bt,h]
  int i = blockIdx.x * blockDim.x + threadIdx.x;
  if (i >= MTOT * NHH) return;
  int h = i % NHH, bt = i / NHH, b = bt / TT;
  float acc = 0.0f;
  for (int d = 0; d < HD; ++d)
    acc += q[(size_t)bt * DD + h * HD + d] * ks[(b * NHH + h) * HD + d];
  z[i] = fmaxf(acc, 1e-6f);
}

__global__ void attn_apply(const float* __restrict__ q, const float* __restrict__ kv,
                           const float* __restrict__ z, float* __restrict__ out) {
  int i = blockIdx.x * blockDim.x + threadIdx.x;       // bt*NH*HD
  if (i >= MTOT * NHH * HD) return;
  int e = i % HD, h = (i / HD) % NHH, bt = i / (HD * NHH), b = bt / TT;
  float acc = 0.0f;
  const float* kvp = kv + (((size_t)(b * NHH + h)) * HD) * HD + e;
  for (int d = 0; d < HD; ++d)
    acc += q[(size_t)bt * DD + h * HD + d] * kvp[(size_t)d * HD];
  out[(size_t)bt * DD + h * HD + e] = acc / z[bt * NHH + h];
}

__global__ void gate_combine(const float* __restrict__ glin,
                             const float* __restrict__ o, float* __restrict__ x, int n) {
  int i = blockIdx.x * blockDim.x + threadIdx.x;
  if (i >= n) return;
  float g = 1.0f / (1.0f + __expf(-glin[i]));
  x[i] = g * o[i] + (1.0f - g) * x[i];
}

__global__ void mamba_scan(float* __restrict__ Bu, const float* __restrict__ log_dt,
                           const float* __restrict__ log_lambda, int layer) {
  int i = blockIdx.x * blockDim.x + threadIdx.x;       // b*S
  if (i >= BBB * SS) return;
  int b = i / SS, s = i % SS;
  float dt  = __expf(log_dt[layer]);
  float lam = -__expf(log_lambda[layer * SS + s]);
  float ld  = lam * dt;
  float acc = 0.0f;
  for (int t = 0; t < TT; ++t) {
    float ldt = fminf(fmaxf((float)t * ld, -30.0f), 0.0f);
    float dec = __expf(ldt);
    float inv = fminf(__expf(-ldt), 1e6f);
    size_t o = ((size_t)b * TT + t) * SS + s;
    acc += Bu[o] * inv;
    Bu[o] = acc * dec;
  }
}

__global__ void add_inplace(float* __restrict__ a, const float* __restrict__ b, int n) {
  int i = blockIdx.x * blockDim.x + threadIdx.x;
  if (i < n) a[i] += b[i];
}

__global__ void silu_mul(float* __restrict__ a, const float* __restrict__ b, int n) {
  int i = blockIdx.x * blockDim.x + threadIdx.x;
  if (i >= n) return;
  float v = a[i];
  a[i] = (v / (1.0f + __expf(-v))) * b[i];
}

__global__ void rmsnorm_k(const float* __restrict__ in, const float* __restrict__ w,
                          float* __restrict__ out, int cols) {
  int row = blockIdx.x;
  __shared__ float red[256];
  float s = 0.0f;
  for (int c = threadIdx.x; c < cols; c += 256) {
    float v = in[(size_t)row * cols + c];
    s += v * v;
  }
  red[threadIdx.x] = s; __syncthreads();
  for (int st = 128; st > 0; st >>= 1) {
    if (threadIdx.x < st) red[threadIdx.x] += red[threadIdx.x + st];
    __syncthreads();
  }
  float inv = rsqrtf(red[0] / (float)cols + 1e-8f);
  for (int c = threadIdx.x; c < cols; c += 256)
    out[(size_t)row * cols + c] = w[c] * in[(size_t)row * cols + c] * inv;
}

__global__ void dot_scalar(const float* __restrict__ x, const float* __restrict__ w,
                           const float* __restrict__ b, float* __restrict__ out, int rows) {
  int bt = blockIdx.x * blockDim.x + threadIdx.x;
  if (bt >= rows) return;
  float acc = 0.0f;
  for (int d = 0; d < DD; ++d) acc += x[(size_t)bt * DD + d] * w[d];
  out[bt] = acc + b[0];
}

__global__ void softmax_rows(const float* __restrict__ in, float* __restrict__ out, int cols) {
  int row = blockIdx.x;
  __shared__ float red[256];
  float m = -1e30f;
  for (int c = threadIdx.x; c < cols; c += 256) m = fmaxf(m, in[(size_t)row * cols + c]);
  red[threadIdx.x] = m; __syncthreads();
  for (int st = 128; st > 0; st >>= 1) {
    if (threadIdx.x < st) red[threadIdx.x] = fmaxf(red[threadIdx.x], red[threadIdx.x + st]);
    __syncthreads();
  }
  m = red[0]; __syncthreads();
  float s = 0.0f;
  for (int c = threadIdx.x; c < cols; c += 256) {
    float e = __expf(in[(size_t)row * cols + c] - m);
    out[(size_t)row * cols + c] = e; s += e;
  }
  red[threadIdx.x] = s; __syncthreads();
  for (int st = 128; st > 0; st >>= 1) {
    if (threadIdx.x < st) red[threadIdx.x] += red[threadIdx.x + st];
    __syncthreads();
  }
  float inv = 1.0f / red[0];
  for (int c = threadIdx.x; c < cols; c += 256) out[(size_t)row * cols + c] *= inv;
}

__global__ void weighted_sum(const float* __restrict__ w, const float* __restrict__ y,
                             float* __restrict__ out) {  // out[b,d] = sum_t w[b,t]*y[bt,d]
  int i = blockIdx.x * blockDim.x + threadIdx.x;
  if (i >= BBB * DD) return;
  int b = i / DD, d = i % DD;
  float acc = 0.0f;
  for (int t = 0; t < TT; ++t)
    acc += w[b * TT + t] * y[((size_t)(b * TT + t)) * DD + d];
  out[i] = acc;
}

__global__ void small_lin(const float* __restrict__ in, const float* __restrict__ W,
                          const float* __restrict__ bias, float* __restrict__ out,
                          int Bn, int N, int K) {       // out[b,n] = in[b,:] . W[n,:] + bias
  int i = blockIdx.x * blockDim.x + threadIdx.x;
  if (i >= Bn * N) return;
  int b = i / N, n = i % N;
  float acc = bias ? bias[n] : 0.0f;
  for (int k = 0; k < K; ++k) acc += in[(size_t)b * K + k] * W[(size_t)n * K + k];
  out[i] = acc;
}

__global__ void node_q(const float* __restrict__ sr, const float* __restrict__ rel,
                       const float* __restrict__ orr, float* __restrict__ qv) {
  int r = blockIdx.x * blockDim.x + threadIdx.x;
  if (r >= RR) return;
  float node = 0.0f;
  for (int b = 0; b < BBB; ++b)
    node += sr[b * RR + r] + rel[b * RR + r] + orr[b * RR + r];
  node *= (1.0f / (float)BBB);
  for (int b = 0; b < BBB; ++b) qv[b * RR + r] = sr[b * RR + r] + node;
}

__global__ void wm_read(const float* __restrict__ init, const float* __restrict__ qv,
                        float* __restrict__ rd) {       // rd[b,r]
  int i = blockIdx.x * blockDim.x + threadIdx.x;
  if (i >= BBB * RR) return;
  int b = i / RR, r = i % RR;
  float sc[8];
  float m = -1e30f;
  for (int s = 0; s < 8; ++s) {
    float a = 0.0f;
    for (int k = 0; k < RR; ++k) a += init[s * RR + k] * qv[b * RR + k];
    sc[s] = a; m = fmaxf(m, a);
  }
  float sum = 0.0f;
  for (int s = 0; s < 8; ++s) { sc[s] = __expf(sc[s] - m); sum += sc[s]; }
  float acc = 0.0f;
  for (int s = 0; s < 8; ++s) acc += (sc[s] / sum) * init[s * RR + r];
  rd[i] = acc;
}

__global__ void add_bcast(float* __restrict__ x, const float* __restrict__ xadd, int n) {
  int i = blockIdx.x * blockDim.x + threadIdx.x;
  if (i >= n) return;
  int d = i % DD, b = i / (TT * DD);
  x[i] += xadd[b * DD + d];
}

// ---------------------------------------------------------------------------
static inline void gemm(const float* A, const float* W, const float* bias, float* C,
                        int M, int N, int K, hipStream_t s) {
  dim3 grid(M / 64, N / 32);
  wmma_gemm_bf16<<<grid, 128, 0, s>>>(A, W, bias, C, M, N, K);
}
#define EW(kern, n, ...) kern<<<((n) + 255) / 256, 256, 0, stream>>>(__VA_ARGS__)

extern "C" void kernel_launch(void* const* d_in, const int* in_sizes, int n_in,
                              void* d_out, int out_size, void* d_ws, size_t ws_size,
                              hipStream_t stream) {
  const int*   ids      = (const int*)  d_in[0];
  const float* emb      = (const float*)d_in[1];
  const float* wq       = (const float*)d_in[2];
  const float* wk       = (const float*)d_in[3];
  const float* wv       = (const float*)d_in[4];
  const float* wo       = (const float*)d_in[5];
  const float* wg       = (const float*)d_in[6];
  const float* bg       = (const float*)d_in[7];
  const float* m_win    = (const float*)d_in[8];   // [L,D,D]
  const float* m_llam   = (const float*)d_in[9];   // [L,S]
  const float* m_B      = (const float*)d_in[10];  // [L,S,D]
  const float* m_C      = (const float*)d_in[11];  // [L,D,S]
  const float* m_ldt    = (const float*)d_in[12];  // [L]
  const float* m_wout   = (const float*)d_in[13];  // [L,D,D]
  const float* m_norm   = (const float*)d_in[14];  // [L,D]
  const float* norm_w   = (const float*)d_in[15];
  const float* f_w1     = (const float*)d_in[16];
  const float* f_b1     = (const float*)d_in[17];
  const float* f_w2     = (const float*)d_in[18];
  const float* f_b2     = (const float*)d_in[19];
  const float* f_w3     = (const float*)d_in[20];
  const float* f_b3     = (const float*)d_in[21];
  const float* wsubj    = (const float*)d_in[22];
  const float* bsubj    = (const float*)d_in[23];
  const float* wact     = (const float*)d_in[24];
  const float* bact     = (const float*)d_in[25];
  const float* wobj     = (const float*)d_in[26];
  const float* bobj     = (const float*)d_in[27];
  const float* c_ws     = (const float*)d_in[28];
  const float* c_bs     = (const float*)d_in[29];
  const float* c_wa     = (const float*)d_in[30];
  const float* c_ba     = (const float*)d_in[31];
  const float* c_wo2    = (const float*)d_in[32];
  const float* c_bo2    = (const float*)d_in[33];
  const float* sp_w     = (const float*)d_in[34];
  const float* sp_b     = (const float*)d_in[35];
  const float* op_w     = (const float*)d_in[36];
  const float* op_b     = (const float*)d_in[37];
  const float* rp_w     = (const float*)d_in[38];
  const float* rp_b     = (const float*)d_in[39];
  const float* wm_init  = (const float*)d_in[40];
  const float* wmb_w    = (const float*)d_in[41];
  const float* wmb_b    = (const float*)d_in[42];
  const float* fnorm    = (const float*)d_in[43];
  const float* lm_head  = (const float*)d_in[44];
  float* out = (float*)d_out;

  // workspace layout (floats)
  float* ws   = (float*)d_ws;
  float* xf   = ws;                         // 4096*768
  float* t1   = xf + (size_t)MTOT * DD;     // 4096*3072
  float* t2   = t1 + (size_t)MTOT * FF;     // 4096*3072
  float* t3   = t2 + (size_t)MTOT * FF;     // 4096*768
  float* t4   = t3 + (size_t)MTOT * DD;     // 4096*768
  float* t5   = t4 + (size_t)MTOT * DD;     // 4096*768
  float* kv   = t5 + (size_t)MTOT * DD;     // B*NH*96*96
  float* ksm  = kv + BBB * NHH * HD * HD;   // B*NH*96
  float* zb   = ksm + BBB * NHH * HD;       // B*T*NH
  float* lg   = zb + MTOT * NHH;            // 3*4096 (s,a,o logits)
  float* sm   = lg + 3 * MTOT;              // 3*4096
  float* cvec = sm + 3 * MTOT;              // 3*B*D (subj, act, obj)
  float* pr   = cvec + 3 * BBB * DD;        // 3*B*R (subj_r, rel, obj_r)
  float* qv   = pr + 3 * BBB * RR;          // B*R
  float* rd   = qv + BBB * RR;              // B*R
  float* xadd = rd + BBB * RR;              // B*D

  const int nXD = MTOT * DD;

  // 1) embedding gather
  EW(embed_gather, nXD, ids, emb, xf, nXD);

  // 2) liquid linear attention
  gemm(xf, wq, nullptr, t1, MTOT, DD, DD, stream);       // q
  gemm(xf, wk, nullptr, t2, MTOT, DD, DD, stream);       // k
  gemm(xf, wv, nullptr, t3, MTOT, DD, DD, stream);       // v
  gemm(xf, wg, bg,      t4, MTOT, DD, DD, stream);       // gate pre-sigmoid
  EW(phi_elu1, nXD, t1, nXD);
  EW(phi_elu1, nXD, t2, nXD);
  EW(rope_k, MTOT * NHH * HALF, t1);
  EW(rope_k, MTOT * NHH * HALF, t2);
  EW(kv_outer, BBB * NHH * HD * HD, t2, t3, kv);
  EW(k_sum, BBB * NHH * HD, t2, ksm);
  EW(z_denom, MTOT * NHH, t1, ksm, zb);
  EW(attn_apply, MTOT * NHH * HD, t1, kv, zb, t5);
  gemm(t5, wo, nullptr, t3, MTOT, DD, DD, stream);       // out proj
  EW(gate_combine, nXD, t4, t3, xf, nXD);

  // 3) 12 mamba blocks
  for (int l = 0; l < LL; ++l) {
    const float* win  = m_win  + (size_t)l * DD * DD;
    const float* wB   = m_B    + (size_t)l * SS * DD;
    const float* wC   = m_C    + (size_t)l * DD * SS;
    const float* wout = m_wout + (size_t)l * DD * DD;
    gemm(xf, win, nullptr, t1, MTOT, DD, DD, stream);    // xi
    gemm(t1, wB,  nullptr, t2, MTOT, SS, DD, stream);    // Bu [4096,64]
    mamba_scan<<<1, 128, 0, stream>>>(t2, m_ldt, m_llam, l);
    gemm(t2, wC,  nullptr, t3, MTOT, DD, SS, stream);    // h @ C^T
    EW(add_inplace, nXD, t3, t1, nXD);                   // xi + s4d
    gemm(t3, wout, nullptr, t4, MTOT, DD, DD, stream);   // xo
    rmsnorm_k<<<MTOT, 256, 0, stream>>>(t4, m_norm + (size_t)l * DD, xf, DD);
  }

  // 4) rmsnorm + FFN
  rmsnorm_k<<<MTOT, 256, 0, stream>>>(xf, norm_w, t3, DD);
  gemm(t3, f_w1, f_b1, t1, MTOT, FF, DD, stream);
  gemm(t3, f_w2, f_b2, t2, MTOT, FF, DD, stream);
  EW(silu_mul, MTOT * FF, t1, t2, MTOT * FF);
  gemm(t1, f_w3, f_b3, xf, MTOT, DD, FF, stream);        // x (no residual)

  // 5) concepts
  gemm(xf, wsubj, bsubj, t1, MTOT, DD, DD, stream);
  gemm(xf, wact,  bact,  t2, MTOT, DD, DD, stream);
  gemm(xf, wobj,  bobj,  t3, MTOT, DD, DD, stream);
  EW(dot_scalar, MTOT, xf, c_ws,  c_bs,  lg,            MTOT);
  EW(dot_scalar, MTOT, xf, c_wa,  c_ba,  lg + MTOT,     MTOT);
  EW(dot_scalar, MTOT, xf, c_wo2, c_bo2, lg + 2 * MTOT, MTOT);
  softmax_rows<<<BBB, 256, 0, stream>>>(lg,            sm,            TT);
  softmax_rows<<<BBB, 256, 0, stream>>>(lg + MTOT,     sm + MTOT,     TT);
  softmax_rows<<<BBB, 256, 0, stream>>>(lg + 2 * MTOT, sm + 2 * MTOT, TT);
  EW(weighted_sum, BBB * DD, sm,            t1, cvec);                // subj
  EW(weighted_sum, BBB * DD, sm + MTOT,     t2, cvec + BBB * DD);     // act
  EW(weighted_sum, BBB * DD, sm + 2 * MTOT, t3, cvec + 2 * BBB * DD); // obj

  // 6) projections + working memory
  EW(small_lin, BBB * RR, cvec,                sp_w, sp_b, pr,                BBB, RR, DD); // subj_r
  EW(small_lin, BBB * RR, cvec + BBB * DD,     rp_w, rp_b, pr + BBB * RR,     BBB, RR, DD); // rel
  EW(small_lin, BBB * RR, cvec + 2 * BBB * DD, op_w, op_b, pr + 2 * BBB * RR, BBB, RR, DD); // obj_r
  EW(node_q, RR, pr, pr + BBB * RR, pr + 2 * BBB * RR, qv);
  EW(wm_read, BBB * RR, wm_init, qv, rd);
  EW(small_lin, BBB * DD, rd, wmb_w, wmb_b, xadd, BBB, DD, RR);
  EW(add_bcast, nXD, xf, xadd, nXD);

  // 7) final norm + lm_head
  rmsnorm_k<<<MTOT, 256, 0, stream>>>(xf, fnorm, t4, DD);
  gemm(t4, lm_head, nullptr, out, MTOT, VV, DD, stream); // [4096, 32000]
}
